// FeatureAttnNet_3753801417417
// MI455X (gfx1250) — matrix operand order repaired and checked by reading, not compiled, not run on hardware
//
#include <hip/hip_runtime.h>

// FeatureAttnNet for MI455X (gfx1250, wave32).
// d_in order: 0 feature[B,256] 1 hidden[B,256] 2 command[B,4] 3 Wq[64,4]
//             4 Wkz[1024,256] 5 Wkh[1024,256] 6 Wvz[1024,256] 7 Wvh[1024,256]
//             8 gammaQ[64] 9 betaQ[64] 10 gammaK[64] 11 betaK[64]
// d_out: s[B,64] f32.
// d_ws: WK bf16 [2048,256] | WV bf16 [2048,256] | stats floats (~2.1 MB total).

#define BATCH   65536
#define EPSBN   1e-5f
#define WSTRIDE 136   // LDS half-tile row stride in bf16 (128 + 8 pad -> conflict-free b128)

typedef __bf16 bf16_t;
typedef bf16_t v16bf __attribute__((ext_vector_type(16)));
typedef float  v8f   __attribute__((ext_vector_type(8)));
typedef float  v4f   __attribute__((ext_vector_type(4)));
typedef int    v4i   __attribute__((ext_vector_type(4)));
typedef unsigned int u32x4 __attribute__((ext_vector_type(4)));

#ifndef __has_builtin
#define __has_builtin(x) 0
#endif
#if __has_builtin(__builtin_amdgcn_global_load_async_to_lds_b128)
#define HAVE_ASYNC_LDS 1
#else
#define HAVE_ASYNC_LDS 0
#endif

typedef __attribute__((address_space(1))) v4i as1_v4i;
typedef __attribute__((address_space(3))) v4i as3_v4i;
#define AS1V4(p) ((as1_v4i*)(void*)(p))
#define AS3V4(p) ((as3_v4i*)(void*)(p))

__device__ __forceinline__ v8f wmma_bf16(v16bf a, v16bf b, v8f c) {
  // D(16x16,f32) = A(16x32,bf16) x B(32x16,bf16) + C
  return __builtin_amdgcn_wmma_f32_16x16x32_bf16(false, a, false, b, (short)0, c, false, false);
}

// A fragment (16x32, M=lane%16): h=0 -> k {kc*32+0..15}; h=1 -> k {kc*32+16..31}.
__device__ __forceinline__ v16bf load_a_frag(const float* row, int kc, int h) {
  const float* p0 = row + kc * 32 + h * 16;
  const float* p1 = p0 + 8;
  v4f a0 = *(const v4f*)p0;
  v4f a1 = *(const v4f*)(p0 + 4);
  v4f b0 = *(const v4f*)p1;
  v4f b1 = *(const v4f*)(p1 + 4);
  union { v16bf v; bf16_t e[16]; } u;
#pragma unroll
  for (int i = 0; i < 4; ++i) {
    u.e[i]      = (bf16_t)a0[i];
    u.e[4 + i]  = (bf16_t)a1[i];
    u.e[8 + i]  = (bf16_t)b0[i];
    u.e[12 + i] = (bf16_t)b1[i];
  }
  return u.v;
}

// B fragment (32x16, N=lane%16): lane half selects k 0..15 / 16..31 -> 32 contiguous bytes.
// kc is local to the staged 128-wide K-half (0..3).
__device__ __forceinline__ v16bf load_b_frag(const bf16_t* tile, int n, int kc, int h) {
  const bf16_t* p = tile + n * WSTRIDE + kc * 32 + h * 16;
  union { v16bf v; u32x4 q[2]; } u;
  u.q[0] = *(const u32x4*)p;
  u.q[1] = *(const u32x4*)(p + 8);
  return u.v;
}

// Stage one 64x128 bf16 weight half-tile (16 KB) into padded LDS.
// Async LDS-DMA when available (ASYNCcnt), else sync copy. 128 threads x 8 b128 chunks.
__device__ __forceinline__ void issue_wtile_half(const bf16_t* __restrict__ gsrc,
                                                 bf16_t* lbuf, int tid) {
#pragma unroll
  for (int i = 0; i < 8; ++i) {
    int c   = tid + i * 128;            // 1024 chunks of 16 B
    int row = c >> 4;
    int cq  = c & 15;
    const bf16_t* g = gsrc + row * 256 + cq * 8;
    bf16_t*       l = lbuf + row * WSTRIDE + cq * 8;
#if HAVE_ASYNC_LDS
    __builtin_amdgcn_global_load_async_to_lds_b128(AS1V4(g), AS3V4(l), 0, 0);
#else
    *(u32x4*)l = *(const u32x4*)g;
#endif
  }
}

__device__ __forceinline__ void wait_async_lds() {
#if HAVE_ASYNC_LDS
  asm volatile("s_wait_asynccnt 0x0" ::: "memory");
#endif
}

// ---------- prep: pack concat weights to bf16 ----------------------------------------------------
__global__ __launch_bounds__(256) void convert_weights(
    const float* __restrict__ Wkz, const float* __restrict__ Wkh,
    const float* __restrict__ Wvz, const float* __restrict__ Wvh,
    bf16_t* __restrict__ WK, bf16_t* __restrict__ WV) {
  int n = blockIdx.x;            // 0..2047
  int k = threadIdx.x;           // 0..255
  int idx = n * 256 + k;
  float kv, vv;
  if (n < 1024) { kv = Wkz[idx];              vv = Wvz[idx]; }
  else          { kv = Wkh[idx - 1024 * 256]; vv = Wvh[idx - 1024 * 256]; }
  WK[idx] = (bf16_t)kv;
  WV[idx] = (bf16_t)vv;
}

// ---------- pass 0: command first/second moments (Q batch-stats are exact from these) ------------
__global__ __launch_bounds__(256) void cmd_stats(const float* __restrict__ command,
                                                 float* __restrict__ csum,
                                                 float* __restrict__ ccsum) {
  int b = blockIdx.x * 256 + threadIdx.x;     // exactly B threads
  v4f c = *(const v4f*)(command + b * 4);
  float vals[20];
#pragma unroll
  for (int i = 0; i < 4; ++i) vals[i] = c[i];
#pragma unroll
  for (int i = 0; i < 4; ++i)
#pragma unroll
    for (int j = 0; j < 4; ++j) vals[4 + i * 4 + j] = c[i] * c[j];
  int lane = threadIdx.x & 31;
#pragma unroll
  for (int i = 0; i < 20; ++i) {
    float v = vals[i];
    for (int m = 1; m < 32; m <<= 1) v += __shfl_xor(v, m, 32);
    if (lane == 0) {
      if (i < 4) atomicAdd(&csum[i], v);
      else       atomicAdd(&ccsum[i - 4], v);
    }
  }
}

// ---------- pass 1: K raw sum / sum-of-squares per d-channel (full K GEMM, nothing stored) -------
__global__ __launch_bounds__(128, 1) void kstats(const float* __restrict__ feature,
                                                 const float* __restrict__ hidden,
                                                 const bf16_t* __restrict__ WK,
                                                 float* __restrict__ ksum,
                                                 float* __restrict__ ksumsq) {
  __shared__ __align__(16) bf16_t wbuf[2][64 * WSTRIDE];
  __shared__ float ksum_s[64], kssq_s[64];
  int tid = threadIdx.x, lane = tid & 31, wid = tid >> 5;
  int r = lane & 15, h = lane >> 4;
  if (tid < 64) { ksum_s[tid] = 0.f; kssq_s[tid] = 0.f; }

  int bbase = blockIdx.x * 64 + wid * 16;
  float sum[4] = {0.f, 0.f, 0.f, 0.f}, ssq[4] = {0.f, 0.f, 0.f, 0.f};
  v8f cacc[4];
  v16bf af[8];

  issue_wtile_half(WK, wbuf[0], tid);                     // prologue: step 0 (f=0, khalf=0)
  for (int s = 0; s < 64; ++s) {                          // 32 f-blocks x 2 K-halves
    int buf = s & 1;
    wait_async_lds();
    __syncthreads();                                      // wbuf[buf] ready; wbuf[buf^1] free
    if (s + 1 < 64) {
      int t1 = s + 1;
      issue_wtile_half(WK + (size_t)(t1 >> 1) * (64 * 256) + (t1 & 1) * 128,
                       wbuf[buf ^ 1], tid);
    }
    int hs = s >> 5, kh = s & 1;
    if ((s & 31) == 0) {                                  // concat half changed -> reload A frags
      const float* src  = hs ? hidden : feature;
      const float* arow = src + (size_t)(bbase + r) * 256;
#pragma unroll
      for (int kc = 0; kc < 8; ++kc) af[kc] = load_a_frag(arow, kc, h);
    }
    const bf16_t* wt = wbuf[buf];
    if (kh == 0) {
#pragma unroll
      for (int t = 0; t < 4; ++t) cacc[t] = (v8f){0.f,0.f,0.f,0.f,0.f,0.f,0.f,0.f};
    }
#pragma unroll
    for (int t = 0; t < 4; ++t) {
      v8f c = cacc[t];
#pragma unroll
      for (int kc = 0; kc < 4; ++kc)
        c = wmma_bf16(af[kh * 4 + kc], load_b_frag(wt, t * 16 + r, kc, h), c);
      cacc[t] = c;
    }
    if (kh == 1) {                                        // full K value ready -> fold moments
#pragma unroll
      for (int t = 0; t < 4; ++t)
#pragma unroll
        for (int j = 0; j < 8; ++j) { float v = cacc[t][j]; sum[t] += v; ssq[t] += v * v; }
    }
  }
#pragma unroll
  for (int t = 0; t < 4; ++t) {
    sum[t] += __shfl_xor(sum[t], 16, 32);                 // fold the two b-halves
    ssq[t] += __shfl_xor(ssq[t], 16, 32);
  }
  if (h == 0) {
#pragma unroll
    for (int t = 0; t < 4; ++t) {
      atomicAdd(&ksum_s[t * 16 + r], sum[t]);
      atomicAdd(&kssq_s[t * 16 + r], ssq[t]);
    }
  }
  __syncthreads();
  if (tid < 64) {
    atomicAdd(&ksum[tid],   ksum_s[tid]);
    atomicAdd(&ksumsq[tid], kssq_s[tid]);
  }
}

// ---------- finalize norm params: q~ = alpha[d]*(Wq[d].c) + beta[d] ------------------------------
__global__ void finalize_stats(const float* __restrict__ Wq,
                               const float* __restrict__ csum, const float* __restrict__ ccsum,
                               const float* __restrict__ ksum, const float* __restrict__ ksumsq,
                               const float* __restrict__ gammaQ, const float* __restrict__ betaQ,
                               const float* __restrict__ gammaK,
                               float* __restrict__ alpha, float* __restrict__ beta_) {
  int d = threadIdx.x;                                    // 64 threads
  v4f wq = *(const v4f*)(Wq + d * 4);
  const float invB = 1.0f / (float)BATCH;
  float mQ = 0.f;
#pragma unroll
  for (int i = 0; i < 4; ++i) mQ += wq[i] * csum[i];
  mQ *= invB;
  float eq2 = 0.f;
#pragma unroll
  for (int i = 0; i < 4; ++i)
#pragma unroll
    for (int j = 0; j < 4; ++j) eq2 += wq[i] * wq[j] * ccsum[i * 4 + j];
  eq2 *= invB;
  float vQ = eq2 - mQ * mQ;
  float qa = gammaQ[d] * rsqrtf(vQ + EPSBN);
  float qb = betaQ[d] - mQ * qa;
  const float invBF = 1.0f / ((float)BATCH * 32.0f);
  float mK = ksum[d] * invBF;
  float vK = ksumsq[d] * invBF - mK * mK;
  float sK = gammaK[d] * rsqrtf(vK + EPSBN);
  alpha[d] = qa * sK;
  beta_[d] = qb * sK;
}

// ---------- pass 2: fused K-GEMM -> scores -> softmax -> V-GEMM -> output ------------------------
__global__ __launch_bounds__(128, 1) void attn_main(const float* __restrict__ feature,
                                                    const float* __restrict__ hidden,
                                                    const float* __restrict__ command,
                                                    const float* __restrict__ Wq,
                                                    const float* __restrict__ alpha,
                                                    const float* __restrict__ beta_,
                                                    const bf16_t* __restrict__ WK,
                                                    const bf16_t* __restrict__ WV,
                                                    float* __restrict__ out) {
  __shared__ __align__(16) bf16_t wbuf[2][64 * WSTRIDE];  // double-buffered weight half-tiles
  __shared__ float sc[32][68];                            // scores [f][b_local] (wave-private cols)

  int tid = threadIdx.x, lane = tid & 31, wid = tid >> 5;
  int r = lane & 15, h = lane >> 4;
  int bbase = blockIdx.x * 64 + wid * 16;

  issue_wtile_half(WK, wbuf[0], tid);                     // prologue of the weight stream

  // ============================ K phase (steps 0..63 over WK) ====================================
  {
    // q~[b,d] at this lane's C-tile coords: b = j + 8h, d = t*16 + r
    float qv[4][8];
    {
      v4f cmd[8];
#pragma unroll
      for (int j = 0; j < 8; ++j)
        cmd[j] = *(const v4f*)(command + (size_t)(bbase + 8 * h + j) * 4);
#pragma unroll
      for (int t = 0; t < 4; ++t) {
        int d = t * 16 + r;
        v4f wq = *(const v4f*)(Wq + d * 4);
        float al = alpha[d], be = beta_[d];
#pragma unroll
        for (int j = 0; j < 8; ++j) {
          float raw = wq[0]*cmd[j][0] + wq[1]*cmd[j][1] + wq[2]*cmd[j][2] + wq[3]*cmd[j][3];
          qv[t][j] = al * raw + be;
        }
      }
    }

    float accK[8];
    v16bf af[8];
    for (int s = 0; s < 64; ++s) {
      int buf = s & 1;
      wait_async_lds();
      __syncthreads();                                    // wbuf[buf] ready; wbuf[buf^1] free
      if (s < 63)
        issue_wtile_half(WK + (size_t)((s + 1) >> 1) * (64 * 256) + ((s + 1) & 1) * 128,
                         wbuf[buf ^ 1], tid);
      else
        issue_wtile_half(WV, wbuf[buf ^ 1], tid);         // prefetch V-phase step 0
      int hs = s >> 5, kh = s & 1, f = s >> 1;
      if ((s & 31) == 0) {                                // concat half changed -> reload A frags
        const float* src  = hs ? hidden : feature;
        const float* arow = src + (size_t)(bbase + r) * 256;
#pragma unroll
        for (int kc = 0; kc < 8; ++kc) af[kc] = load_a_frag(arow, kc, h);
      }
      const bf16_t* wt = wbuf[buf];
      if (kh == 0) {
#pragma unroll
        for (int j = 0; j < 8; ++j) accK[j] = 0.f;
      }
#pragma unroll
      for (int t = 0; t < 4; ++t) {
        v8f c = (v8f){0.f,0.f,0.f,0.f,0.f,0.f,0.f,0.f};
#pragma unroll
        for (int kc = 0; kc < 4; ++kc)
          c = wmma_bf16(af[kh * 4 + kc], load_b_frag(wt, t * 16 + r, kc, h), c);
#pragma unroll
        for (int j = 0; j < 8; ++j) accK[j] += c[j] * qv[t][j];
      }
      if (kh == 1) {
        for (int m = 1; m < 16; m <<= 1)
#pragma unroll
          for (int j = 0; j < 8; ++j) accK[j] += __shfl_xor(accK[j], m, 32);
        if (r == 0)
#pragma unroll
          for (int j = 0; j < 8; ++j) sc[f][wid * 16 + 8 * h + j] = accK[j] * 0.125f;
      }
    }
  }

  // ============================ softmax (wave-private sc columns) ================================
  // Overlaps the in-flight async prefetch of the first V-phase weight tile.
  if (lane < 16) {
    int col = wid * 16 + lane;
    float v[32], mx = -1e30f;
#pragma unroll
    for (int ff = 0; ff < 32; ++ff) { v[ff] = sc[ff][col]; mx = fmaxf(mx, v[ff]); }
    float sum = 0.f;
#pragma unroll
    for (int ff = 0; ff < 32; ++ff) { v[ff] = __expf(v[ff] - mx); sum += v[ff]; }
    float inv = 1.0f / sum;
#pragma unroll
    for (int ff = 0; ff < 32; ++ff) sc[ff][col] = v[ff] * inv;
  }

  // ============================ V phase (steps 0..63 over WV) ====================================
  {
    v8f oacc[4];
#pragma unroll
    for (int t = 0; t < 4; ++t) oacc[t] = (v8f){0.f,0.f,0.f,0.f,0.f,0.f,0.f,0.f};
    v16bf af[8];
    for (int s = 0; s < 64; ++s) {
      int buf = s & 1;                                    // parity continues from K loop
      wait_async_lds();
      __syncthreads();
      if (s < 63)
        issue_wtile_half(WV + (size_t)((s + 1) >> 1) * (64 * 256) + ((s + 1) & 1) * 128,
                         wbuf[buf ^ 1], tid);
      int hs = s >> 5, kh = s & 1, f = s >> 1;
      if ((s & 31) == 0) {
        const float* src  = hs ? hidden : feature;
        const float* arow = src + (size_t)(bbase + r) * 256;
#pragma unroll
        for (int kc = 0; kc < 8; ++kc) af[kc] = load_a_frag(arow, kc, h);
      }
      const bf16_t* wt = wbuf[buf];
      float aw[8];
#pragma unroll
      for (int j = 0; j < 8; ++j) aw[j] = sc[f][wid * 16 + 8 * h + j];
#pragma unroll
      for (int t = 0; t < 4; ++t) {
        v8f c = (v8f){0.f,0.f,0.f,0.f,0.f,0.f,0.f,0.f};
#pragma unroll
        for (int kc = 0; kc < 4; ++kc)
          c = wmma_bf16(af[kh * 4 + kc], load_b_frag(wt, t * 16 + r, kc, h), c);
#pragma unroll
        for (int j = 0; j < 8; ++j) oacc[t][j] += aw[j] * c[j];
      }
    }

#pragma unroll
    for (int t = 0; t < 4; ++t)
#pragma unroll
      for (int j = 0; j < 8; ++j)
        out[(size_t)(bbase + 8 * h + j) * 64 + t * 16 + r] = oacc[t][j];
  }
}

// -------------------------------------------------------------------------------------------------
extern "C" void kernel_launch(void* const* d_in, const int* in_sizes, int n_in,
                              void* d_out, int out_size, void* d_ws, size_t ws_size,
                              hipStream_t stream) {
  (void)in_sizes; (void)n_in; (void)out_size; (void)ws_size;
  const float* feature = (const float*)d_in[0];
  const float* hidden  = (const float*)d_in[1];
  const float* command = (const float*)d_in[2];
  const float* Wq      = (const float*)d_in[3];
  const float* Wkz     = (const float*)d_in[4];
  const float* Wkh     = (const float*)d_in[5];
  const float* Wvz     = (const float*)d_in[6];
  const float* Wvh     = (const float*)d_in[7];
  const float* gammaQ  = (const float*)d_in[8];
  const float* betaQ   = (const float*)d_in[9];
  const float* gammaK  = (const float*)d_in[10];
  // betaK (d_in[11]) provably cancels under softmax -> unused.

  bf16_t* WK = (bf16_t*)d_ws;
  bf16_t* WV = WK + (size_t)2048 * 256;
  float* stats  = (float*)((char*)d_ws + (size_t)2 * 2048 * 256 * sizeof(bf16_t));
  float* csum   = stats;        // 4
  float* ccsum  = stats + 4;    // 16
  float* ksum   = stats + 20;   // 64
  float* ksumsq = stats + 84;   // 64
  float* alpha  = stats + 148;  // 64
  float* beta_  = stats + 212;  // 64

  (void)hipMemsetAsync(stats, 0, 148 * sizeof(float), stream);

  convert_weights<<<2048, 256, 0, stream>>>(Wkz, Wkh, Wvz, Wvh, WK, WV);
  cmd_stats<<<BATCH / 256, 256, 0, stream>>>(command, csum, ccsum);
  kstats<<<BATCH / 64, 128, 0, stream>>>(feature, hidden, WK, ksum, ksumsq);
  finalize_stats<<<1, 64, 0, stream>>>(Wq, csum, ccsum, ksum, ksumsq,
                                       gammaQ, betaQ, gammaK, alpha, beta_);
  attn_main<<<BATCH / 64, 128, 0, stream>>>(feature, hidden, command, Wq,
                                            alpha, beta_, WK, WV, (float*)d_out);
}
